// SupervisedGATEncoder_13262859010451
// MI455X (gfx1250) — compile-verified, matrix operand order checked by reading, not run on hardware
//
#include <hip/hip_runtime.h>
#include <hip/hip_bf16.h>

#define NODES   50000
#define EDGES   800000
#define ETOT    (EDGES + NODES)   // + self loops
#define IN_DIM  128
#define HIDC    64
#define EMB     32
#define HEADS   4
#define C0      (HEADS * HIDC)    // 256
#define NEG_SLOPE 0.2f
#define BN_EPS  1e-5f
#define MAX_K   256               // max GEMM reduction dim (layer 1)
#define PANELW  128               // max column panel per block

typedef __attribute__((ext_vector_type(16))) _Float16 v16h;
typedef __attribute__((ext_vector_type(8)))  float    v8f;
typedef __attribute__((ext_vector_type(4)))  unsigned v4u;
typedef __attribute__((ext_vector_type(8)))  int      v8i;
typedef __attribute__((ext_vector_type(4)))  int      v4i;

#if defined(__gfx1250__) && __has_builtin(__builtin_amdgcn_tensor_load_to_lds) && \
    __has_builtin(__builtin_amdgcn_s_wait_tensorcnt)
#define HAVE_TDM 1
#else
#define HAVE_TDM 0
#endif

// ---------------------------------------------------------------------------
// WMMA GEMM: C[M,Ncols] = A[M,K] * B[K,Ncols], fp32 in memory, f16 fragments,
// f32 accumulation via V_WMMA_F32_16X16X32_F16.
// Block = 256 threads (8 waves): rows [16*bx, 16*bx+16), cols
// [128*by, 128*by + panelW). B panel (K x panelW f32) staged into LDS by the
// Tensor Data Mover (one descriptor, wave 0, S_WAIT_TENSORCNT); A strip
// (16 x K f32) staged cooperatively with coalesced loads. Each wave then
// computes one 16x16 tile per panel column group entirely out of LDS.
// Fragment layouts per ISA 7.12.2 (16-bit A 16x32 / B 32x16 / f32 C 16x16).
// ---------------------------------------------------------------------------
__global__ void wmma_gemm_tiled(const float* __restrict__ A,
                                const float* __restrict__ B,
                                float* __restrict__ C,
                                int M, int Ncols, int K, int panelW) {
  __shared__ float Bs[MAX_K * PANELW / 2];   // 64 KB: [K][panelW], K*panelW <= 16384
  __shared__ float As[16 * MAX_K];           // 16 KB: [16][K]

  const int tid  = threadIdx.x;
  const int lane = tid & 31;
  const int wave = tid >> 5;
  const int m0   = blockIdx.x * 16;
  const int col0 = blockIdx.y * PANELW;

  // ---- stage B panel: Tensor Data Mover (TENSORcnt) or cooperative loads --
#if HAVE_TDM
  if (wave == 0) {
    const unsigned lds_off = (unsigned)(unsigned long long)(void*)&Bs[0];
    const unsigned long long gaddr =
        (unsigned long long)(const void*)(B + col0);
    // D# group 0: count=1, lds_addr, 57-bit global_addr, type=2 ("image")
    v4u g0 = { 1u, lds_off,
               (unsigned)(gaddr & 0xFFFFFFFFu),
               (unsigned)((gaddr >> 32) & 0x01FFFFFFu) | 0x80000000u };
    // D# group 1: data_size=4B (code 2); tensor_dim0=Ncols; tensor_dim1=K;
    // tile_dim0=panelW; tile_dim1=K; tensor_dim0_stride=Ncols
    const unsigned td0 = (unsigned)Ncols, td1 = (unsigned)K;
    v8i g1 = { (int)(2u << 16),
               (int)((td0 & 0xFFFFu) << 16),
               (int)(((td0 >> 16) & 0xFFFFu) | ((td1 & 0xFFFFu) << 16)),
               (int)(((td1 >> 16) & 0xFFFFu) | (((unsigned)panelW & 0xFFFFu) << 16)),
               (int)((unsigned)K & 0xFFFFu),
               (int)(unsigned)Ncols,   // dim0 stride (elements)
               0, 0 };
    v4i gz4 = { 0, 0, 0, 0 };
    v8i gz8 = { 0, 0, 0, 0, 0, 0, 0, 0 };
    __builtin_amdgcn_tensor_load_to_lds(g0, g1, gz4, gz4, gz8, 0);
    __builtin_amdgcn_s_wait_tensorcnt(0);
  }
#else
  for (int idx = tid; idx < K * panelW; idx += 256) {
    const int k = idx / panelW, c = idx % panelW;
    Bs[idx] = B[k * Ncols + col0 + c];
  }
#endif

  // ---- stage A strip (16 rows x K), coalesced along K ---------------------
  for (int idx = tid; idx < 16 * K; idx += 256) {
    const int r = idx / K, k = idx % K;
    As[idx] = A[(m0 + r) * K + k];
  }
  __syncthreads();

  // ---- compute: one 16x16 tile per wave out of LDS ------------------------
  const int n0 = wave * 16;                 // column offset inside panel
  if (n0 < panelW) {                        // wave-uniform: EXEC stays all-ones
    const int hs = lane >> 4;
    const int lr = lane & 15;
    v8f acc = {};
    for (int kb = 0; kb < K; kb += 32) {
      v16h af, bf;
#pragma unroll
      for (int v = 0; v < 8; ++v) {
        const int koff = 2 * v + (v >= 4 ? 8 : 0) + 8 * hs;
        const int k = kb + koff;
        af[2 * v]     = (_Float16)As[lr * K + k];
        af[2 * v + 1] = (_Float16)As[lr * K + k + 1];
        bf[2 * v]     = (_Float16)Bs[k * panelW + n0 + lr];
        bf[2 * v + 1] = (_Float16)Bs[(k + 1) * panelW + n0 + lr];
      }
      acc = __builtin_amdgcn_wmma_f32_16x16x32_f16(
          /*neg_a=*/false, af, /*neg_b=*/false, bf,
          /*c_mod=*/(short)0, acc, /*reuse_a=*/false, /*reuse_b=*/false);
    }
#pragma unroll
    for (int r = 0; r < 8; ++r) {
      C[(m0 + r + 8 * hs) * Ncols + (col0 + n0 + lr)] = acc[r];
    }
  }
}

// ---------------------------------------------------------------------------
// Per-node attention logits
// ---------------------------------------------------------------------------
__global__ void alpha0_kernel(const float* __restrict__ H0,
                              const float* __restrict__ a_src,
                              const float* __restrict__ a_dst,
                              float* __restrict__ as, float* __restrict__ ad) {
  const int idx = blockIdx.x * blockDim.x + threadIdx.x;   // node*HEADS + h
  if (idx >= NODES * HEADS) return;
  const int node = idx >> 2, h = idx & 3;
  const float* row = H0 + node * C0 + h * HIDC;
  float s = 0.f, d = 0.f;
#pragma unroll 8
  for (int c = 0; c < HIDC; ++c) {
    const float v = row[c];
    s += v * a_src[h * HIDC + c];
    d += v * a_dst[h * HIDC + c];
  }
  as[idx] = s; ad[idx] = d;
}

__global__ void alpha1_kernel(const float* __restrict__ H1,
                              const float* __restrict__ a_src,
                              const float* __restrict__ a_dst,
                              float* __restrict__ as, float* __restrict__ ad) {
  const int n = blockIdx.x * blockDim.x + threadIdx.x;
  if (n >= NODES) return;
  const float* row = H1 + n * EMB;
  float s = 0.f, d = 0.f;
#pragma unroll
  for (int c = 0; c < EMB; ++c) {
    const float v = row[c];
    s += v * a_src[c];
    d += v * a_dst[c];
  }
  as[n] = s; ad[n] = d;
}

// Monotonic float <-> ordered-u32 mapping for atomicMax-based segment max.
__device__ __forceinline__ unsigned f2ord(float f) {
  unsigned u = __float_as_uint(f);
  return (u & 0x80000000u) ? ~u : (u | 0x80000000u);
}
__device__ __forceinline__ float ord2f(unsigned k) {
  return __uint_as_float((k & 0x80000000u) ? (k & 0x7FFFFFFFu) : ~k);
}

__device__ __forceinline__ void edge_sd(int i, const int* __restrict__ src,
                                        const int* __restrict__ dst,
                                        int& s, int& d) {
  if (i < EDGES) { s = src[i]; d = dst[i]; }
  else           { s = i - EDGES; d = i - EDGES; }   // self-loops
}

// ---- layer 0 edge pipeline (4 heads) --------------------------------------
__global__ void edge_max0(const int* __restrict__ src, const int* __restrict__ dst,
                          const float* __restrict__ as, const float* __restrict__ ad,
                          unsigned* __restrict__ mmax) {
  const int i = blockIdx.x * blockDim.x + threadIdx.x;
  if (i >= ETOT) return;
  int s, d; edge_sd(i, src, dst, s, d);
#pragma unroll
  for (int h = 0; h < HEADS; ++h) {
    float e = as[s * HEADS + h] + ad[d * HEADS + h];
    e = e > 0.f ? e : NEG_SLOPE * e;
    atomicMax(&mmax[d * HEADS + h], f2ord(e));
  }
}

__global__ void edge_expsum0(const int* __restrict__ src, const int* __restrict__ dst,
                             const float* __restrict__ as, const float* __restrict__ ad,
                             const unsigned* __restrict__ mmax,
                             float* __restrict__ ex, float* __restrict__ denom) {
  const int i = blockIdx.x * blockDim.x + threadIdx.x;
  if (i >= ETOT) return;
  int s, d; edge_sd(i, src, dst, s, d);
#pragma unroll
  for (int h = 0; h < HEADS; ++h) {
    float e = as[s * HEADS + h] + ad[d * HEADS + h];
    e = e > 0.f ? e : NEG_SLOPE * e;
    const float v = __expf(e - ord2f(mmax[d * HEADS + h]));
    ex[i * HEADS + h] = v;
    atomicAdd(&denom[d * HEADS + h], v);
  }
}

// one 256-thread block per edge: coalesced 256-channel gather + scatter-add
__global__ void edge_agg0(const int* __restrict__ src, const int* __restrict__ dst,
                          const float* __restrict__ H0, const float* __restrict__ ex,
                          const float* __restrict__ denom, float* __restrict__ agg) {
  const int i = blockIdx.x;
  const int c = threadIdx.x;
  int s, d; edge_sd(i, src, dst, s, d);
  const int h = c >> 6;
  const float w = ex[i * HEADS + h] / denom[d * HEADS + h];
  atomicAdd(&agg[d * C0 + c], w * H0[s * C0 + c]);
}

// bias + BatchNorm(eval) + ReLU -> h1
__global__ void bn_relu(const float* __restrict__ agg, const float* __restrict__ b0,
                        const float* __restrict__ g, const float* __restrict__ be,
                        const float* __restrict__ mu, const float* __restrict__ var,
                        float* __restrict__ h1) {
  const int idx = blockIdx.x * blockDim.x + threadIdx.x;
  if (idx >= NODES * C0) return;
  const int c = idx & (C0 - 1);
  float v = agg[idx] + b0[c];
  v = (v - mu[c]) * (g[c] * rsqrtf(var[c] + BN_EPS)) + be[c];
  h1[idx] = fmaxf(v, 0.f);
}

// ---- layer 1 edge pipeline (1 head) ---------------------------------------
__global__ void edge_max1(const int* __restrict__ src, const int* __restrict__ dst,
                          const float* __restrict__ as, const float* __restrict__ ad,
                          unsigned* __restrict__ mmax) {
  const int i = blockIdx.x * blockDim.x + threadIdx.x;
  if (i >= ETOT) return;
  int s, d; edge_sd(i, src, dst, s, d);
  float e = as[s] + ad[d];
  e = e > 0.f ? e : NEG_SLOPE * e;
  atomicMax(&mmax[d], f2ord(e));
}

__global__ void edge_expsum1(const int* __restrict__ src, const int* __restrict__ dst,
                             const float* __restrict__ as, const float* __restrict__ ad,
                             const unsigned* __restrict__ mmax,
                             float* __restrict__ ex, float* __restrict__ denom) {
  const int i = blockIdx.x * blockDim.x + threadIdx.x;
  if (i >= ETOT) return;
  int s, d; edge_sd(i, src, dst, s, d);
  float e = as[s] + ad[d];
  e = e > 0.f ? e : NEG_SLOPE * e;
  const float v = __expf(e - ord2f(mmax[d]));
  ex[i] = v;
  atomicAdd(&denom[d], v);
}

__global__ void edge_agg1(const int* __restrict__ src, const int* __restrict__ dst,
                          const float* __restrict__ H1, const float* __restrict__ ex,
                          const float* __restrict__ denom, float* __restrict__ agg) {
  const long long t = (long long)blockIdx.x * blockDim.x + threadIdx.x;
  if (t >= (long long)ETOT * EMB) return;
  const int i = (int)(t >> 5);
  const int c = (int)(t & (EMB - 1));
  int s, d; edge_sd(i, src, dst, s, d);
  const float w = ex[i] / denom[d];
  atomicAdd(&agg[d * EMB + c], w * H1[s * EMB + c]);
}

// node embeddings out + global-mean accumulation
__global__ void finalize_nodes(const float* __restrict__ agg, const float* __restrict__ b1,
                               float* __restrict__ out, float* __restrict__ gsum) {
  const int idx = blockIdx.x * blockDim.x + threadIdx.x;
  if (idx >= NODES * EMB) return;
  const int c = idx & (EMB - 1);
  const float v = agg[idx] + b1[c];
  out[idx] = v;
  atomicAdd(&gsum[c], v);
}

// tiny MLP head: one block of 64 threads
__global__ void head_kernel(const float* __restrict__ gsum,
                            const float* __restrict__ hW1, const float* __restrict__ hb1,
                            const float* __restrict__ hW2, const float* __restrict__ hb2,
                            float* __restrict__ pred) {
  __shared__ float ge[EMB];
  __shared__ float hid[HIDC];
  const int t = threadIdx.x;
  if (t < EMB) ge[t] = gsum[t] / (float)NODES;
  __syncthreads();
  float a = hb1[t];
#pragma unroll
  for (int c = 0; c < EMB; ++c) a += ge[c] * hW1[c * HIDC + t];
  hid[t] = fmaxf(a, 0.f);
  __syncthreads();
  if (t == 0) {
    float p = hb2[0];
#pragma unroll
    for (int j = 0; j < HIDC; ++j) p += hid[j] * hW2[j];
    pred[0] = p;
  }
}

extern "C" void kernel_launch(void* const* d_in, const int* in_sizes, int n_in,
                              void* d_out, int out_size, void* d_ws, size_t ws_size,
                              hipStream_t stream) {
  (void)in_sizes; (void)n_in; (void)out_size; (void)ws_size;
  const float* x       = (const float*)d_in[0];
  const int*   eidx    = (const int*)d_in[1];
  const float* W0      = (const float*)d_in[2];
  const float* a_src0  = (const float*)d_in[3];
  const float* a_dst0  = (const float*)d_in[4];
  const float* b0      = (const float*)d_in[5];
  const float* bn_g    = (const float*)d_in[6];
  const float* bn_b    = (const float*)d_in[7];
  const float* bn_m    = (const float*)d_in[8];
  const float* bn_v    = (const float*)d_in[9];
  const float* W1      = (const float*)d_in[10];
  const float* a_src1  = (const float*)d_in[11];
  const float* a_dst1  = (const float*)d_in[12];
  const float* b1      = (const float*)d_in[13];
  const float* hW1     = (const float*)d_in[14];
  const float* hb1     = (const float*)d_in[15];
  const float* hW2     = (const float*)d_in[16];
  const float* hb2     = (const float*)d_in[17];

  const int* src = eidx;
  const int* dst = eidx + EDGES;
  float* out = (float*)d_out;

  // ---- workspace layout (buffers reused across phases) --------------------
  float*    H0   = (float*)d_ws;              // [N,256]; reused as h1 after BN
  float*    agg0 = H0 + (size_t)NODES * C0;   // [N,256]; reused as H1 (layer 1 GEMM out)
  float*    ex0  = agg0 + (size_t)NODES * C0; // [ETOT,4]; reused as ex1 ([ETOT])
  float*    as0  = ex0 + (size_t)ETOT * HEADS;
  float*    ad0  = as0 + (size_t)NODES * HEADS;
  unsigned* mx0  = (unsigned*)(ad0 + (size_t)NODES * HEADS);
  float*    dn0  = (float*)(mx0 + (size_t)NODES * HEADS);
  float*    as1  = dn0 + (size_t)NODES * HEADS;
  float*    ad1  = as1 + NODES;
  unsigned* mx1  = (unsigned*)(ad1 + NODES);
  float*    dn1  = (float*)(mx1 + NODES);
  float*    agg1 = dn1 + NODES;               // [N,32]
  float*    gsum = agg1 + (size_t)NODES * EMB;

  const int EB = 256;
  const int eblocks = (ETOT + EB - 1) / EB;

  // ---- layer 0: GEMM H0 = x @ W0 (TDM-staged LDS, WMMA f16, f32 accum) ----
  wmma_gemm_tiled<<<dim3(NODES / 16, C0 / PANELW), 256, 0, stream>>>(
      x, W0, H0, NODES, C0, IN_DIM, PANELW);
  alpha0_kernel<<<(NODES * HEADS + 255) / 256, 256, 0, stream>>>(
      H0, a_src0, a_dst0, as0, ad0);

  (void)hipMemsetAsync(mx0, 0, (size_t)NODES * HEADS * sizeof(unsigned), stream);
  (void)hipMemsetAsync(dn0, 0, (size_t)NODES * HEADS * sizeof(float), stream);
  edge_max0<<<eblocks, EB, 0, stream>>>(src, dst, as0, ad0, mx0);
  edge_expsum0<<<eblocks, EB, 0, stream>>>(src, dst, as0, ad0, mx0, ex0, dn0);
  (void)hipMemsetAsync(agg0, 0, (size_t)NODES * C0 * sizeof(float), stream);
  edge_agg0<<<ETOT, C0, 0, stream>>>(src, dst, H0, ex0, dn0, agg0);

  bn_relu<<<(NODES * C0 + 255) / 256, 256, 0, stream>>>(
      agg0, b0, bn_g, bn_b, bn_m, bn_v, H0);      // h1 overwrites H0 buffer

  // ---- layer 1: GEMM H1 = h1 @ W1 ----------------------------------------
  float* H1 = agg0;                               // reuse buffer
  wmma_gemm_tiled<<<dim3(NODES / 16, 1), 256, 0, stream>>>(
      H0, W1, H1, NODES, EMB, C0, EMB);
  alpha1_kernel<<<(NODES + 255) / 256, 256, 0, stream>>>(
      H1, a_src1, a_dst1, as1, ad1);

  (void)hipMemsetAsync(mx1, 0, (size_t)NODES * sizeof(unsigned), stream);
  (void)hipMemsetAsync(dn1, 0, (size_t)NODES * sizeof(float), stream);
  float* ex1 = ex0;                               // reuse buffer
  edge_max1<<<eblocks, EB, 0, stream>>>(src, dst, as1, ad1, mx1);
  edge_expsum1<<<eblocks, EB, 0, stream>>>(src, dst, as1, ad1, mx1, ex1, dn1);
  (void)hipMemsetAsync(agg1, 0, (size_t)NODES * EMB * sizeof(float), stream);
  {
    const long long tot = (long long)ETOT * EMB;
    edge_agg1<<<(unsigned)((tot + 255) / 256), 256, 0, stream>>>(
        src, dst, H1, ex1, dn1, agg1);
  }

  (void)hipMemsetAsync(gsum, 0, EMB * sizeof(float), stream);
  finalize_nodes<<<(NODES * EMB + 255) / 256, 256, 0, stream>>>(agg1, b1, out, gsum);
  head_kernel<<<1, HIDC, 0, stream>>>(gsum, hW1, hb1, hW2, hb2, out + (size_t)NODES * EMB);
}
